// GNNModel_29188597744083
// MI455X (gfx1250) — compile-verified, matrix operand order checked by reading, not run on hardware
//
#include <hip/hip_runtime.h>

// ---------------------------------------------------------------------------
// GNN (2-branch shared-weight GCN) for MI455X / gfx1250, wave32.
// N=100000 nodes, C_IN=64, C_H=128, C_OUT=16, E=1600000 edges, L=3 layers.
// Dense GEMMs via v_wmma_f32_16x16x32_bf16 with fragment-ordered LDS tiles;
// edge aggregation via f32 global atomics (lands in the 192MB L2).
// ---------------------------------------------------------------------------

typedef __attribute__((ext_vector_type(16))) __bf16 v16bf;
typedef __attribute__((ext_vector_type(8)))  float  v8f;

#define GNN_N 100000
#define GNN_E 1600000
#define C_H   128

// Inverse of the WMMA 16-bit operand k-map.
// Fragment element i of lane-half h holds K offset kr = 8*h + (i&7) + 16*(i>>3).
// Given kr in [0,32): hi = kr>>3 -> half = hi&1, i = (kr&7) + ((hi>>1)<<3).
__device__ __forceinline__ void kmap_inv(int kr, int& half, int& i)
{
    const int hi = kr >> 3;
    half = hi & 1;
    i    = (kr & 7) + ((hi >> 1) << 3);
}

// ---------------------------------------------------------------------------
// Per-node GEMM: C[N x 128] = act(A[N x K] @ W[K x 128] (+ bias))
// Block = 256 threads = 8 waves; block tile = 32 rows x 128 cols.
// Wave w owns column group [16w,16w+16) and both 16-row tiles (B reuse x2).
// A and W are staged into LDS in WMMA per-lane fragment order so the inner
// loop is pure ds_load_b128 + v_wmma (no scalar LDS loads, no b16 packing).
// ---------------------------------------------------------------------------
template <int K, bool RELU, bool BIAS>
__global__ __launch_bounds__(256) void gemm_node_kernel(
    const float* __restrict__ A, const float* __restrict__ W,
    const float* __restrict__ bias, float* __restrict__ C, int N)
{
    constexpr int KB   = K / 32;                 // k-blocks per row
    constexpr int KSH  = (K == 128) ? 7 : 6;     // log2(K)
    constexpr int RT   = 2;                      // row tiles per block
    // Fragment layout: frag[(tile*KB + kb)*32 + lane][16] contiguous bf16.
    __shared__ __align__(32) __bf16 Af[RT * KB * 512];
    __shared__ __align__(32) __bf16 Wf[8  * KB * 512];

    const int tid  = threadIdx.x;
    const int row0 = blockIdx.x * 32;

    // ---- Stage A (32 x K), source-linear (coalesced), fragment-order dest.
    for (int idx = tid; idx < 32 * K; idx += 256) {
        const int k  = idx & (K - 1);
        const int r  = idx >> KSH;
        const float v = A[(size_t)(row0 + r) * K + k];
        const int rt = r >> 4, rl = r & 15;
        const int kb = k >> 5, kr = k & 31;
        int half, i;  kmap_inv(kr, half, i);
        const int lane = rl + (half << 4);
        Af[((rt * KB + kb) * 32 + lane) * 16 + i] = (__bf16)v;
    }
    // ---- Stage W (K x 128), source-linear (coalesced), fragment-order dest.
    for (int idx = tid; idx < K * 128; idx += 256) {
        const int n = idx & 127;
        const int k = idx >> 7;
        const float v = W[idx];
        const int wg = n >> 4, nl = n & 15;
        const int kb = k >> 5, kr = k & 31;
        int half, i;  kmap_inv(kr, half, i);
        const int lane = nl + (half << 4);
        Wf[((wg * KB + kb) * 32 + lane) * 16 + i] = (__bf16)v;
    }
    __syncthreads();

    const int lane = tid & 31;
    const int wv   = tid >> 5;
    const int nc   = (wv << 4) + (lane & 15);    // output column

    v8f acc0 = {}, acc1 = {};
#pragma unroll
    for (int kb = 0; kb < KB; ++kb) {
        const v16bf b  = *(const v16bf*)&Wf[((wv * KB + kb) * 32 + lane) * 16];
        const v16bf a0 = *(const v16bf*)&Af[((0  * KB + kb) * 32 + lane) * 16];
        const v16bf a1 = *(const v16bf*)&Af[((1  * KB + kb) * 32 + lane) * 16];
        acc0 = __builtin_amdgcn_wmma_f32_16x16x32_bf16(
            false, a0, false, b, (short)0, acc0, false, false);
        acc1 = __builtin_amdgcn_wmma_f32_16x16x32_bf16(
            false, a1, false, b, (short)0, acc1, false, false);
    }

    // D layout: lanes 0-15 hold M=r in VGPR r; lanes 16-31 hold M=8+r.
    float bv = 0.0f;
    if (BIAS) bv = bias[nc];
    const int moff = (lane >> 4) << 3;
#pragma unroll
    for (int r = 0; r < 8; ++r) {
        float v0 = acc0[r], v1 = acc1[r];
        if (BIAS) { v0 += bv; v1 += bv; }
        if (RELU) { v0 = fmaxf(v0, 0.0f); v1 = fmaxf(v1, 0.0f); }
        C[(size_t)(row0 +      moff + r) * 128 + nc] = v0;
        C[(size_t)(row0 + 16 + moff + r) * 128 + nc] = v1;
    }
}

// ---------------------------------------------------------------------------
// Degree count (in-degree via dst) as int atomics.
// ---------------------------------------------------------------------------
__global__ void degree_kernel(const int* __restrict__ dst, int* __restrict__ deg, int E)
{
    const int e = blockIdx.x * blockDim.x + threadIdx.x;
    if (e < E) atomicAdd(&deg[dst[e]], 1);
}

__global__ void dis_kernel(const int* __restrict__ deg, float* __restrict__ dis, int N)
{
    const int i = blockIdx.x * blockDim.x + threadIdx.x;
    if (i < N) dis[i] = rsqrtf((float)deg[i] + 1.0f);
}

// ---------------------------------------------------------------------------
// Edge scatter: agg[dst] += t[src] * dis[src]*dis[dst].
// One wave32 per edge, each lane moves a float4 (128 cols total).
// ---------------------------------------------------------------------------
__global__ __launch_bounds__(256) void scatter_edges_kernel(
    const int* __restrict__ src, const int* __restrict__ dst,
    const float* __restrict__ dis, const float* __restrict__ t,
    float* __restrict__ agg, int E)
{
    const int gid  = blockIdx.x * blockDim.x + threadIdx.x;
    const int e    = gid >> 5;
    if (e >= E) return;
    const int lane = gid & 31;
    const int s = src[e];
    const int d = dst[e];
    const float coef = dis[s] * dis[d];

    const float4 m = *reinterpret_cast<const float4*>(t + (size_t)s * 128 + lane * 4);
    float* ap = agg + (size_t)d * 128 + lane * 4;
    atomicAdd(ap + 0, m.x * coef);
    atomicAdd(ap + 1, m.y * coef);
    atomicAdd(ap + 2, m.z * coef);
    atomicAdd(ap + 3, m.w * coef);
}

// ---------------------------------------------------------------------------
// Fused: v = agg + t*dis^2 + conv_b ; LayerNorm(v)*g + b ; ReLU.
// One wave32 per node, 4 columns per lane, shuffle-based reductions.
// ---------------------------------------------------------------------------
__global__ __launch_bounds__(256) void combine_ln_kernel(
    const float* __restrict__ t, const float* __restrict__ agg,
    const float* __restrict__ dis, const float* __restrict__ cb,
    const float* __restrict__ g, const float* __restrict__ bb,
    float* __restrict__ out, int N)
{
    const int gid  = blockIdx.x * blockDim.x + threadIdx.x;
    const int node = gid >> 5;
    if (node >= N) return;
    const int lane = gid & 31;
    const int c0   = lane * 4;

    const float di   = dis[node];
    const float self = di * di;
    const size_t base = (size_t)node * 128 + c0;

    const float4 tv = *reinterpret_cast<const float4*>(t + base);
    const float4 av = *reinterpret_cast<const float4*>(agg + base);

    float v[4];
    v[0] = av.x + tv.x * self + cb[c0 + 0];
    v[1] = av.y + tv.y * self + cb[c0 + 1];
    v[2] = av.z + tv.z * self + cb[c0 + 2];
    v[3] = av.w + tv.w * self + cb[c0 + 3];

    float s = v[0] + v[1] + v[2] + v[3];
#pragma unroll
    for (int off = 16; off > 0; off >>= 1) s += __shfl_xor(s, off, 32);
    const float mean = s * 0.0078125f;   // 1/128

    float d0 = v[0] - mean, d1 = v[1] - mean, d2 = v[2] - mean, d3 = v[3] - mean;
    float sq = d0 * d0 + d1 * d1 + d2 * d2 + d3 * d3;
#pragma unroll
    for (int off = 16; off > 0; off >>= 1) sq += __shfl_xor(sq, off, 32);
    const float inv = rsqrtf(sq * 0.0078125f + 1e-5f);

    float4 o;
    o.x = fmaxf(d0 * inv * g[c0 + 0] + bb[c0 + 0], 0.0f);
    o.y = fmaxf(d1 * inv * g[c0 + 1] + bb[c0 + 1], 0.0f);
    o.z = fmaxf(d2 * inv * g[c0 + 2] + bb[c0 + 2], 0.0f);
    o.w = fmaxf(d3 * inv * g[c0 + 3] + bb[c0 + 3], 0.0f);
    *reinterpret_cast<float4*>(out + base) = o;
}

// ---------------------------------------------------------------------------
// Final: out[N x 16] = concat(f1, f2)[N x 256] @ out_w[256 x 16] + out_b.
// One thread per (node, out-col); W is 16 KB and stays cache-resident.
// ---------------------------------------------------------------------------
__global__ __launch_bounds__(256) void final_gemm_kernel(
    const float* __restrict__ f1, const float* __restrict__ f2,
    const float* __restrict__ W, const float* __restrict__ b,
    float* __restrict__ out, int N)
{
    const int gid  = blockIdx.x * blockDim.x + threadIdx.x;
    const int node = gid >> 4;
    if (node >= N) return;
    const int o = gid & 15;

    const float* h1 = f1 + (size_t)node * 128;
    const float* h2 = f2 + (size_t)node * 128;
    float acc = b[o];
#pragma unroll 8
    for (int k = 0; k < 128; ++k) acc += h1[k] * W[k * 16 + o];
#pragma unroll 8
    for (int k = 0; k < 128; ++k) acc += h2[k] * W[(128 + k) * 16 + o];
    out[(size_t)node * 16 + o] = acc;
}

// ---------------------------------------------------------------------------
// Host launcher.
// ---------------------------------------------------------------------------
extern "C" void kernel_launch(void* const* d_in, const int* in_sizes, int n_in,
                              void* d_out, int out_size, void* d_ws, size_t ws_size,
                              hipStream_t stream)
{
    (void)in_sizes; (void)n_in; (void)out_size; (void)ws_size;

    const float* x     = (const float*)d_in[0];
    const int*   ei[2] = {(const int*)d_in[1], (const int*)d_in[2]};
    const float* enc_w = (const float*)d_in[3];
    const float* enc_b = (const float*)d_in[4];
    const float* conv_w[3] = {(const float*)d_in[5],  (const float*)d_in[9],  (const float*)d_in[13]};
    const float* conv_b[3] = {(const float*)d_in[6],  (const float*)d_in[10], (const float*)d_in[14]};
    const float* ln_g[3]   = {(const float*)d_in[7],  (const float*)d_in[11], (const float*)d_in[15]};
    const float* ln_b[3]   = {(const float*)d_in[8],  (const float*)d_in[12], (const float*)d_in[16]};
    const float* out_w = (const float*)d_in[17];
    const float* out_b = (const float*)d_in[18];

    const int N = GNN_N, E = GNN_E;
    const size_t feat = (size_t)N * C_H;

    float* f1  = (float*)d_ws;          // branch-1 features (persist)
    float* f2  = f1  + feat;            // branch-2 features (persist)
    float* t   = f2  + feat;            // post-GEMM h = x @ W
    float* agg = t   + feat;            // scatter accumulator
    float* dis = agg + feat;            // deg^{-1/2}
    int*   deg = (int*)(dis + N);

    const dim3 blk(256);
    const int gemm_grid    = N / 32;                 // 3125
    const int edge_grid    = (E + 255) / 256;
    const int node_grid    = (N + 255) / 256;
    const int scatter_grid = (int)(((size_t)E * 32 + 255) / 256);   // 200000
    const int wave_grid    = (int)(((size_t)N * 32 + 255) / 256);
    const int final_grid   = (int)(((size_t)N * 16 + 255) / 256);

    for (int br = 0; br < 2; ++br) {
        const int* src = ei[br];
        const int* dst = ei[br] + E;
        float* f = (br == 0) ? f1 : f2;

        // Degree + symmetric normalization coefficients.
        hipMemsetAsync(deg, 0, (size_t)N * sizeof(int), stream);
        degree_kernel<<<edge_grid, blk, 0, stream>>>(dst, deg, E);
        dis_kernel<<<node_grid, blk, 0, stream>>>(deg, dis, N);

        // Encoder: h = relu(x @ enc_w + enc_b)
        gemm_node_kernel<64, true, true><<<gemm_grid, blk, 0, stream>>>(x, enc_w, enc_b, f, N);

        for (int l = 0; l < 3; ++l) {
            // t = h @ conv_w
            gemm_node_kernel<128, false, false><<<gemm_grid, blk, 0, stream>>>(f, conv_w[l], nullptr, t, N);
            // agg = scatter_add(t[src] * dis[src]*dis[dst] -> dst)
            hipMemsetAsync(agg, 0, feat * sizeof(float), stream);
            scatter_edges_kernel<<<scatter_grid, blk, 0, stream>>>(src, dst, dis, t, agg, E);
            // h = relu(LN(agg + t*dis^2 + conv_b) * g + b)
            combine_ln_kernel<<<wave_grid, blk, 0, stream>>>(t, agg, dis, conv_b[l], ln_g[l], ln_b[l], f, N);
        }
    }

    final_gemm_kernel<<<final_grid, blk, 0, stream>>>(f1, f2, out_w, out_b, (float*)d_out, N);
}
